// GATLayer_31026843746678
// MI455X (gfx1250) — matrix operand order repaired and checked by reading, not compiled, not run on hardware
//
#include <hip/hip_runtime.h>
#include <hip/hip_bf16.h>

#define ALPHA 0.2f
#define N_NODES 8192
#define F_OUT 64

typedef __attribute__((ext_vector_type(16))) _Float16 v16h;
typedef __attribute__((ext_vector_type(8)))  _Float16 v8h;
typedef __attribute__((ext_vector_type(8)))  float    v8f;
typedef __attribute__((ext_vector_type(2)))  float    v2f;

// Order-preserving uint encoding of float for atomicMax-based global max.
__device__ __forceinline__ unsigned fenc(float f) {
    unsigned u = __float_as_uint(f);
    return (u & 0x80000000u) ? ~u : (u | 0x80000000u);
}
__device__ __forceinline__ float fdec(unsigned e) {
    return (e & 0x80000000u) ? __uint_as_float(e & 0x7fffffffu)
                             : __uint_as_float(~e);
}

__global__ void gat_init(unsigned* gmax_enc) { *gmax_enc = 0u; }

// ---------------------------------------------------------------------------
// Phase 1: Wh = x@W via v_wmma_f32_16x16x4_f32. One wave per 16-row tile,
// covering all 64 output columns (4 n-tiles). Writes:
//   WhT  : f16 [64][8192]  (transposed, feeds phase-2 B operand contiguously)
//   Wh1  : f32 [8192]      (Wh @ a[:64])
//   Wh2  : f32 [8192]      (Wh @ a[64:])
//   gmax : encoded max_j Wh2[j]  (for softmax upper bound)
// ---------------------------------------------------------------------------
__global__ __launch_bounds__(128) void gat_gemm1(
    const float* __restrict__ x, const float* __restrict__ W,
    const float* __restrict__ a,
    _Float16* __restrict__ WhT, float* __restrict__ Wh1,
    float* __restrict__ Wh2, unsigned* __restrict__ gmax_enc)
{
    const int wave = (blockIdx.x * blockDim.x + threadIdx.x) >> 5;  // 0..511
    const int lane = threadIdx.x & 31;
    const int l15  = lane & 15;
    const int half = lane >> 4;      // 0: K+0/1, M base 0 ; 1: K+2/3, M base 8
    const int i0   = wave * 16;

    v8f c0 = {}, c1 = {}, c2 = {}, c3 = {};
    const float* xrow = x + (size_t)(i0 + l15) * 128 + half * 2;

    for (int k = 0; k < 128; k += 4) {
        v2f A = { xrow[k], xrow[k + 1] };            // K = k+2*half, +1
        const int kk = k + half * 2;
        const float* w0 = W + (size_t)kk * F_OUT + l15;
        const float* w1 = W + (size_t)(kk + 1) * F_OUT + l15;
        v2f B0 = { w0[0],  w1[0]  };
        v2f B1 = { w0[16], w1[16] };
        v2f B2 = { w0[32], w1[32] };
        v2f B3 = { w0[48], w1[48] };
        c0 = __builtin_amdgcn_wmma_f32_16x16x4_f32(false, A, false, B0, (short)0, c0, false, false);
        c1 = __builtin_amdgcn_wmma_f32_16x16x4_f32(false, A, false, B1, (short)0, c1, false, false);
        c2 = __builtin_amdgcn_wmma_f32_16x16x4_f32(false, A, false, B2, (short)0, c2, false, false);
        c3 = __builtin_amdgcn_wmma_f32_16x16x4_f32(false, A, false, B3, (short)0, c3, false, false);
    }

    // C layout: reg r, lanes 0-15 -> M=r, N=lane ; lanes 16-31 -> M=r+8, N=lane-16
    const int moff = half * 8;

    // Store WhT (f16, transposed): 8 consecutive M per lane -> one 16B store per n-tile.
    v8h h0, h1, h2, h3;
    #pragma unroll
    for (int r = 0; r < 8; ++r) {
        h0[r] = (_Float16)c0[r]; h1[r] = (_Float16)c1[r];
        h2[r] = (_Float16)c2[r]; h3[r] = (_Float16)c3[r];
    }
    const size_t mbase = (size_t)i0 + moff;
    *(v8h*)(WhT + (size_t)(l15      ) * N_NODES + mbase) = h0;
    *(v8h*)(WhT + (size_t)(l15 + 16 ) * N_NODES + mbase) = h1;
    *(v8h*)(WhT + (size_t)(l15 + 32 ) * N_NODES + mbase) = h2;
    *(v8h*)(WhT + (size_t)(l15 + 48 ) * N_NODES + mbase) = h3;

    // Wh1/Wh2: weighted reduce over N (across the 16 lanes of each half).
    const float a10 = a[l15],       a11 = a[l15 + 16],
                a12 = a[l15 + 32],  a13 = a[l15 + 48];
    const float a20 = a[64 + l15],      a21 = a[64 + l15 + 16],
                a22 = a[64 + l15 + 32], a23 = a[64 + l15 + 48];
    #pragma unroll
    for (int r = 0; r < 8; ++r) {
        float p1 = c0[r]*a10 + c1[r]*a11 + c2[r]*a12 + c3[r]*a13;
        float p2 = c0[r]*a20 + c1[r]*a21 + c2[r]*a22 + c3[r]*a23;
        #pragma unroll
        for (int mk = 1; mk < 16; mk <<= 1) {
            p1 += __shfl_xor(p1, mk, 32);
            p2 += __shfl_xor(p2, mk, 32);
        }
        if (l15 == 0) {
            Wh1[i0 + moff + r] = p1;
            Wh2[i0 + moff + r] = p2;
            atomicMax(gmax_enc, fenc(p2));
        }
    }
}

// ---------------------------------------------------------------------------
// Phase 2: fused masked-softmax attention + (P @ Wh) with f16 WMMA.
// 512 blocks (one 16-row i-tile each), 8 waves/block striding over 256
// 32-wide j-tiles. Softmax uses fixed per-row bound m̂_i = leaky(Wh1_i + gmax),
// which is exactly equivalent (shift invariance) and needs no rescaling.
// ---------------------------------------------------------------------------
__global__ __launch_bounds__(256) void gat_attn(
    const int* __restrict__ adj, const _Float16* __restrict__ WhT,
    const float* __restrict__ Wh1, const float* __restrict__ Wh2,
    const unsigned* __restrict__ gmax_enc, float* __restrict__ out)
{
    __shared__ float sAcc[8][16 * F_OUT];
    __shared__ float sSum[8][16];
    __shared__ float sWh1[16], sM[16];

    const int tid  = threadIdx.x;
    const int wave = tid >> 5;
    const int lane = tid & 31;
    const int i0   = blockIdx.x * 16;

    if (tid < 16) {
        const float gmax = fdec(*gmax_enc);
        const float w1 = Wh1[i0 + tid];
        sWh1[tid] = w1;
        const float t = w1 + gmax;
        sM[tid] = (t > 0.f) ? t : ALPHA * t;    // upper bound on row max of e
    }
    __syncthreads();

    const int m     = lane & 15;       // row within tile (A-matrix M)
    const int half  = lane >> 4;
    const int kbase = half * 8;        // A layout: lanes 16-31 handle K+8
    const float wh1m = sWh1[m];
    const float mrow = sM[m];

    v8f c0 = {}, c1 = {}, c2 = {}, c3 = {};
    float psum = 0.f;

    const int* adjrow = adj + (size_t)(i0 + m) * N_NODES;

    for (int jt = wave; jt < N_NODES / 32; jt += 8) {
        const int j0 = jt * 32;

        const int4* ap = (const int4*)(adjrow + j0 + kbase);
        int4 q0 = ap[0], q1 = ap[1];       // K = kbase+0..7
        int4 q4 = ap[4], q5 = ap[5];       // K = 16+kbase+0..7
        const float4* wp = (const float4*)(Wh2 + j0 + kbase);
        float4 f0 = wp[0], f1 = wp[1], f4 = wp[4], f5 = wp[5];

        v16h A;
        #define GAT_P(idx, adjv, whv) {                                   \
            float t = wh1m + (whv);                                       \
            t = (t > 0.f) ? t : ALPHA * t;                                \
            float p = ((adjv) > 0) ? __expf(t - mrow) : 0.f;              \
            psum += p;                                                    \
            A[idx] = (_Float16)p; }
        GAT_P(0,  q0.x, f0.x)  GAT_P(1,  q0.y, f0.y)
        GAT_P(2,  q0.z, f0.z)  GAT_P(3,  q0.w, f0.w)
        GAT_P(4,  q1.x, f1.x)  GAT_P(5,  q1.y, f1.y)
        GAT_P(6,  q1.z, f1.z)  GAT_P(7,  q1.w, f1.w)
        GAT_P(8,  q4.x, f4.x)  GAT_P(9,  q4.y, f4.y)
        GAT_P(10, q4.z, f4.z)  GAT_P(11, q4.w, f4.w)
        GAT_P(12, q5.x, f5.x)  GAT_P(13, q5.y, f5.y)
        GAT_P(14, q5.z, f5.z)  GAT_P(15, q5.w, f5.w)
        #undef GAT_P

        // B tiles: lanes 0-15 -> K=j0..j0+7, j0+16..23 of row N=m ; lanes 16-31 -> K=j0+8..15, j0+24..31
        const size_t bcol = (size_t)j0 + half * 8;
        #define GAT_LB(p) __builtin_shufflevector(*(const v8h*)(p), *(const v8h*)((p) + 16), 0,1,2,3,4,5,6,7,8,9,10,11,12,13,14,15)
        v16h B0 = GAT_LB(WhT + (size_t)(m      ) * N_NODES + bcol);
        v16h B1 = GAT_LB(WhT + (size_t)(m + 16 ) * N_NODES + bcol);
        v16h B2 = GAT_LB(WhT + (size_t)(m + 32 ) * N_NODES + bcol);
        v16h B3 = GAT_LB(WhT + (size_t)(m + 48 ) * N_NODES + bcol);
        #undef GAT_LB

        c0 = __builtin_amdgcn_wmma_f32_16x16x32_f16(false, A, false, B0, (short)0, c0, false, false);
        c1 = __builtin_amdgcn_wmma_f32_16x16x32_f16(false, A, false, B1, (short)0, c1, false, false);
        c2 = __builtin_amdgcn_wmma_f32_16x16x32_f16(false, A, false, B2, (short)0, c2, false, false);
        c3 = __builtin_amdgcn_wmma_f32_16x16x32_f16(false, A, false, B3, (short)0, c3, false, false);
    }

    // Combine K-halves of the row sum, dump per-wave partials to LDS.
    psum += __shfl_xor(psum, 16, 32);
    const int moff = half * 8;
    float* acc = sAcc[wave];
    #pragma unroll
    for (int r = 0; r < 8; ++r) {
        const int mm = r + moff;
        acc[mm * F_OUT + m     ] = c0[r];
        acc[mm * F_OUT + m + 16] = c1[r];
        acc[mm * F_OUT + m + 32] = c2[r];
        acc[mm * F_OUT + m + 48] = c3[r];
    }
    if (half == 0) sSum[wave][m] = psum;
    __syncthreads();

    // Cross-wave reduce, softmax divide, ELU, store.
    for (int e = tid; e < 16 * F_OUT; e += 256) {
        const int mm = e >> 6;
        float v = 0.f, s = 0.f;
        #pragma unroll
        for (int w = 0; w < 8; ++w) { v += sAcc[w][e]; s += sSum[w][mm]; }
        float o = v / s;
        o = (o > 0.f) ? o : (__expf(o) - 1.f);   // ELU
        out[(size_t)(i0 + mm) * F_OUT + (e & 63)] = o;
    }
}

// ---------------------------------------------------------------------------
extern "C" void kernel_launch(void* const* d_in, const int* in_sizes, int n_in,
                              void* d_out, int out_size, void* d_ws, size_t ws_size,
                              hipStream_t stream) {
    const float* x   = (const float*)d_in[0];   // [8192,128]
    const int*   adj = (const int*)d_in[1];     // [8192,8192]
    const float* W   = (const float*)d_in[2];   // [128,64]
    const float* a   = (const float*)d_in[3];   // [128,1]
    float* out = (float*)d_out;                 // [8192,64]

    char* ws = (char*)d_ws;
    _Float16* WhT = (_Float16*)ws;                              // 64*8192*2 = 1 MiB
    float*    Wh1 = (float*)(ws + (size_t)N_NODES * F_OUT * 2);          // 32 KiB
    float*    Wh2 = (float*)(ws + (size_t)N_NODES * F_OUT * 2 + 32768);  // 32 KiB
    unsigned* gmx = (unsigned*)(ws + (size_t)N_NODES * F_OUT * 2 + 65536);

    gat_init<<<1, 1, 0, stream>>>(gmx);
    gat_gemm1<<<N_NODES / 16 / 4, 128, 0, stream>>>(x, W, a, WhT, Wh1, Wh2, gmx);
    gat_attn<<<N_NODES / 16, 256, 0, stream>>>(adj, WhT, Wh1, Wh2, gmx, out);
}
